// SelfAttentionV2_48722109006488
// MI455X (gfx1250) — matrix operand order repaired and checked by reading, not compiled, not run on hardware
//
#include <hip/hip_runtime.h>

#define DIM    1024
#define SEQ    4096
#define BATCH  4
#define ROWS   (BATCH * SEQ)      // 16384
#define NE     (3 * DIM)          // 3072

typedef __attribute__((ext_vector_type(16))) _Float16 v16h;
typedef __attribute__((ext_vector_type(8)))  _Float16 v8h;
typedef __attribute__((ext_vector_type(8)))  float    v8f;
typedef __attribute__((ext_vector_type(4)))  float    v4f;

__device__ __forceinline__ v8f wmma_f16f32(v16h a, v16h b, v8f c) {
  // D = A(16x32 f16) * B(32x16 f16) + C(16x16 f32)
  return __builtin_amdgcn_wmma_f32_16x16x32_f16(false, a, false, b, (short)0, c,
                                                false, false);
}

// Load A fragment (16x32, f16) from contiguous f16 row data.
// Lane (l&15) = row. halves[0..7] at k0+8*hi, halves[8..15] at k0+16+8*hi.
__device__ __forceinline__ v16h load_a_f16(const _Float16* rowp, int k0, int hi) {
  const _Float16* p = rowp + k0 + 8 * hi;
  v8h lo = *(const v8h*)(p);
  v8h hh = *(const v8h*)(p + 16);
  v16h a;
#pragma unroll
  for (int i = 0; i < 8; ++i) { a[i] = lo[i]; a[8 + i] = hh[i]; }
  return a;
}

// Load B fragment (32x16, f16) from contiguous f16 column data.
// Lane (l&15) = col. halves[0..7] at k0+8*hi, halves[8..15] at k0+16+8*hi.
__device__ __forceinline__ v16h load_b_f16(const _Float16* colp, int k0, int hi) {
  const _Float16* p = colp + k0 + 8 * hi;
  v8h lo = *(const v8h*)(p);
  v8h hh = *(const v8h*)(p + 16);
  v16h b;
#pragma unroll
  for (int i = 0; i < 8; ++i) { b[i] = lo[i]; b[8 + i] = hh[i]; }
  return b;
}

// ---------------------------------------------------------------------------
// Kernel 0: bulk f32 -> f16 conversion (memory-bound one-time pass)
// ---------------------------------------------------------------------------
__global__ __launch_bounds__(256) void cvt_f32_f16_kernel(
    const float* __restrict__ src, _Float16* __restrict__ dst, int n) {
  const int i = (blockIdx.x * 256 + threadIdx.x) * 8;
  if (i < n) {
    v4f a = *(const v4f*)(src + i);
    v4f b = *(const v4f*)(src + i + 4);
    v8h o;
#pragma unroll
    for (int j = 0; j < 4; ++j) {
      o[j]     = (_Float16)a[j];
      o[4 + j] = (_Float16)b[j];
    }
    *(v8h*)(dst + i) = o;
  }
}

// ---------------------------------------------------------------------------
// Shared epilogue for the QKV kernels: add bias, route one 16-row group of a
// 16(N)-wide accumulator frag to Q / K / V^T as f16.
// ---------------------------------------------------------------------------
__device__ __forceinline__ void qkv_store_frag(
    v8f acc, int r0, int e, int hi, float bv, _Float16* __restrict__ Qh,
    _Float16* __restrict__ Kh, _Float16* __restrict__ Vt) {
  if (e < DIM) {
#pragma unroll
    for (int r = 0; r < 8; ++r) {
      const int row = r0 + r + 8 * hi;
      Qh[(size_t)row * DIM + e] = (_Float16)(acc[r] + bv);
    }
  } else if (e < 2 * DIM) {
    const int e1 = e - DIM;
#pragma unroll
    for (int r = 0; r < 8; ++r) {
      const int row = r0 + r + 8 * hi;
      Kh[(size_t)row * DIM + e1] = (_Float16)(acc[r] + bv);
    }
  } else {
    const int e2 = e - 2 * DIM;
#pragma unroll
    for (int r = 0; r < 8; ++r) {
      const int row = r0 + r + 8 * hi;
      const int b   = row >> 12;          // row / SEQ
      const int s   = row & (SEQ - 1);    // row % SEQ
      Vt[((size_t)(b * DIM + e2)) * SEQ + s] = (_Float16)(acc[r] + bv);
    }
  }
}

// ---------------------------------------------------------------------------
// Kernel 1 (fast path): QKV = Xh @ Wh^T + b, all-f16 operands.
// One wave computes a 32(M) x 64(N) tile: 2 A-frags x 4 B-frags = 8 WMMAs
// per 32-deep k-step; 24 B/lane of loads per WMMA.
// ---------------------------------------------------------------------------
__global__ __launch_bounds__(32) void qkv_f16_kernel(
    const _Float16* __restrict__ Xh, const _Float16* __restrict__ Wh,
    const float* __restrict__ bias, _Float16* __restrict__ Qh,
    _Float16* __restrict__ Kh, _Float16* __restrict__ Vt) {
  const int lane = threadIdx.x;
  const int hi   = lane >> 4;
  const int lid  = lane & 15;
  const int e0   = blockIdx.x * 64;   // output-feature tile
  const int r0   = blockIdx.y * 32;   // row tile (flat over B*S)

  v8f acc[2][4];
  v8f z = {};
#pragma unroll
  for (int g = 0; g < 2; ++g)
#pragma unroll
    for (int n = 0; n < 4; ++n) acc[g][n] = z;

  const _Float16* xrow0 = Xh + (size_t)(r0 + lid) * DIM;
  const _Float16* xrow1 = Xh + (size_t)(r0 + 16 + lid) * DIM;

  for (int k0 = 0; k0 < DIM; k0 += 32) {
    v16h a0 = load_a_f16(xrow0, k0, hi);
    v16h a1 = load_a_f16(xrow1, k0, hi);
#pragma unroll
    for (int n = 0; n < 4; ++n) {
      v16h bfr = load_b_f16(Wh + (size_t)(e0 + n * 16 + lid) * DIM, k0, hi);
      acc[0][n] = wmma_f16f32(a0, bfr, acc[0][n]);
      acc[1][n] = wmma_f16f32(a1, bfr, acc[1][n]);
    }
  }

#pragma unroll
  for (int n = 0; n < 4; ++n) {
    const int e    = e0 + n * 16 + lid;
    const float bv = bias[e];
    qkv_store_frag(acc[0][n], r0, e, hi, bv, Qh, Kh, Vt);
    qkv_store_frag(acc[1][n], r0 + 16, e, hi, bv, Qh, Kh, Vt);
  }
}

// ---------------------------------------------------------------------------
// Kernel 1 (fallback, f32 inputs): one wave computes 16x64; used only when the
// workspace is too small to hold the f16 copies of X and W.
// ---------------------------------------------------------------------------
__global__ __launch_bounds__(32) void qkv_f32_kernel(
    const float* __restrict__ X, const float* __restrict__ W,
    const float* __restrict__ bias, _Float16* __restrict__ Qh,
    _Float16* __restrict__ Kh, _Float16* __restrict__ Vt) {
  const int lane = threadIdx.x;
  const int hi   = lane >> 4;
  const int lid  = lane & 15;
  const int e0   = blockIdx.x * 64;
  const int r0   = blockIdx.y * 16;

  v8f acc[4];
  v8f z = {};
#pragma unroll
  for (int n = 0; n < 4; ++n) acc[n] = z;

  const float* xrow = X + (size_t)(r0 + lid) * DIM;

  for (int k0 = 0; k0 < DIM; k0 += 32) {
    v16h a;
    {
      const float* p = xrow + k0 + 8 * hi;
      v4f f0 = *(const v4f*)(p);
      v4f f1 = *(const v4f*)(p + 4);
      v4f f2 = *(const v4f*)(p + 16);
      v4f f3 = *(const v4f*)(p + 20);
#pragma unroll
      for (int i = 0; i < 4; ++i) {
        a[i]      = (_Float16)f0[i];
        a[4 + i]  = (_Float16)f1[i];
        a[8 + i]  = (_Float16)f2[i];
        a[12 + i] = (_Float16)f3[i];
      }
    }
#pragma unroll
    for (int n = 0; n < 4; ++n) {
      const float* wp = W + (size_t)(e0 + n * 16 + lid) * DIM + k0 + 8 * hi;
      v4f g0 = *(const v4f*)(wp);
      v4f g1 = *(const v4f*)(wp + 4);
      v4f g2 = *(const v4f*)(wp + 16);
      v4f g3 = *(const v4f*)(wp + 20);
      v16h bfr;
#pragma unroll
      for (int i = 0; i < 4; ++i) {
        bfr[i]      = (_Float16)g0[i];
        bfr[4 + i]  = (_Float16)g1[i];
        bfr[8 + i]  = (_Float16)g2[i];
        bfr[12 + i] = (_Float16)g3[i];
      }
      acc[n] = wmma_f16f32(a, bfr, acc[n]);
    }
  }

#pragma unroll
  for (int n = 0; n < 4; ++n) {
    const int e    = e0 + n * 16 + lid;
    const float bv = bias[e];
    qkv_store_frag(acc[n], r0, e, hi, bv, Qh, Kh, Vt);
  }
}

// ---------------------------------------------------------------------------
// Kernel 2: flash attention. Block = 128 threads (4 waves) per 16-row q-tile.
// Wave w computes scores for kv slice w*16 within each 64-kv chunk (shared via
// LDS as f16 P), then accumulates P@V into its private 256-column output
// slice (16x256 f32 = 128 VGPRs/lane).
// ---------------------------------------------------------------------------
__global__ __launch_bounds__(128) void attn_kernel(
    const _Float16* __restrict__ Qh, const _Float16* __restrict__ Kh,
    const _Float16* __restrict__ Vt, float* __restrict__ Out) {
  __shared__ _Float16 sQ[16 * DIM];   // 32 KB q-tile
  __shared__ _Float16 sP[16 * 64];    // 2 KB probs chunk
  __shared__ float sWmax[4][16];
  __shared__ float sWsum[4][16];
  __shared__ float sM[16];
  __shared__ float sL[16];

  const int tid  = threadIdx.x;
  const int wave = tid >> 5;
  const int lane = tid & 31;
  const int hi   = lane >> 4;
  const int lid  = lane & 15;

  const int tile = blockIdx.x;
  const int b    = tile >> 8;           // 256 q-tiles per batch
  const int q0   = (tile & 255) * 16;

  // Stage Q tile into LDS (rows are contiguous in Qh).
  {
    const v8h* src = (const v8h*)(Qh + (size_t)(b * SEQ + q0) * DIM);
    v8h* dst = (v8h*)sQ;
#pragma unroll 4
    for (int i = tid; i < (16 * DIM) / 8; i += 128) dst[i] = src[i];
  }
  if (tid < 16) { sM[tid] = -3.0e38f; sL[tid] = 0.0f; }
  __syncthreads();

  const int d0 = wave * 256;            // this wave's output column slice
  v8f acc[16];
  v8f z = {};
#pragma unroll
  for (int nt = 0; nt < 16; ++nt) acc[nt] = z;

  const _Float16* Kbase = Kh + (size_t)b * SEQ * DIM;
  const _Float16* Vbase = Vt + (size_t)b * DIM * SEQ;
  const float scale = 0.03125f;         // 1/sqrt(1024)

  for (int kv0 = 0; kv0 < SEQ; kv0 += 64) {
    if (kv0 + 64 < SEQ) {
      __builtin_prefetch(Kbase + (size_t)(kv0 + 64 + wave * 16 + lid) * DIM, 0, 1);
      __builtin_prefetch(Vbase + (size_t)(d0 + lid) * SEQ + kv0 + 64, 0, 1);
    }

    // ---- S = Q @ K^T for kv slice [kv0 + wave*16, +16) over K=1024 ----
    v8f s = z;
    const _Float16* krow = Kbase + (size_t)(kv0 + wave * 16 + lid) * DIM;
    const _Float16* qrow = sQ + lid * DIM;
    for (int k0 = 0; k0 < DIM; k0 += 32) {
      v16h a  = load_a_f16(qrow, k0, hi);     // from LDS
      v16h bb = load_b_f16(krow, k0, hi);     // from global (L2-resident)
      s = wmma_f16f32(a, bb, s);
    }

    // scale + per-row max (rows r+8*hi live across this 16-lane half)
    float sv[8], rmax[8];
#pragma unroll
    for (int r = 0; r < 8; ++r) {
      sv[r] = s[r] * scale;
      float m = sv[r];
#pragma unroll
      for (int x = 1; x < 16; x <<= 1) m = fmaxf(m, __shfl_xor(m, x, 32));
      rmax[r] = m;
    }
    if (lid == 0) {
#pragma unroll
      for (int r = 0; r < 8; ++r) sWmax[wave][r + 8 * hi] = rmax[r];
    }
    __syncthreads();

    // new running max + rescale factor (read sM before anyone updates it)
    float mnew[8], alpha[8];
#pragma unroll
    for (int r = 0; r < 8; ++r) {
      const int row = r + 8 * hi;
      float mc = fmaxf(fmaxf(sWmax[0][row], sWmax[1][row]),
                       fmaxf(sWmax[2][row], sWmax[3][row]));
      float mn = fmaxf(sM[row], mc);
      mnew[r]  = mn;
      alpha[r] = __expf(sM[row] - mn);
    }

    // P = exp(S - m_new): write f16 chunk to LDS + per-row partial sums
    float rsum[8];
#pragma unroll
    for (int r = 0; r < 8; ++r) {
      float p = __expf(sv[r] - mnew[r]);
      sP[(r + 8 * hi) * 64 + wave * 16 + lid] = (_Float16)p;
      float t = p;
#pragma unroll
      for (int x = 1; x < 16; x <<= 1) t += __shfl_xor(t, x, 32);
      rsum[r] = t;
    }
    if (lid == 0) {
#pragma unroll
      for (int r = 0; r < 8; ++r) sWsum[wave][r + 8 * hi] = rsum[r];
    }
    __syncthreads();

    // one lane per row updates running stats
    if (tid < 16) {
      float mc = fmaxf(fmaxf(sWmax[0][tid], sWmax[1][tid]),
                       fmaxf(sWmax[2][tid], sWmax[3][tid]));
      float mn = fmaxf(sM[tid], mc);
      float a  = __expf(sM[tid] - mn);
      sL[tid] = a * sL[tid] + sWsum[0][tid] + sWsum[1][tid] +
                sWsum[2][tid] + sWsum[3][tid];
      sM[tid] = mn;
    }
    __syncthreads();

    // ---- O_slice = alpha * O_slice + P(16x64) @ V(64x256) ----
    v16h pa[2];
#pragma unroll
    for (int kf = 0; kf < 2; ++kf) pa[kf] = load_a_f16(sP + lid * 64, kf * 32, hi);

#pragma unroll
    for (int nt = 0; nt < 16; ++nt) {
#pragma unroll
      for (int r = 0; r < 8; ++r) acc[nt][r] *= alpha[r];
      const _Float16* vcol = Vbase + (size_t)(d0 + nt * 16 + lid) * SEQ + kv0;
#pragma unroll
      for (int kf = 0; kf < 2; ++kf) {
        v16h vb = load_b_f16(vcol, kf * 32, hi);
        acc[nt] = wmma_f16f32(pa[kf], vb, acc[nt]);
      }
    }
  }

  // Epilogue: normalize by running sum, store f32 output.
  float linv[8];
#pragma unroll
  for (int r = 0; r < 8; ++r) linv[r] = 1.0f / sL[r + 8 * hi];
#pragma unroll
  for (int nt = 0; nt < 16; ++nt) {
    const int dcol = d0 + nt * 16 + lid;
#pragma unroll
    for (int r = 0; r < 8; ++r) {
      const int row = q0 + r + 8 * hi;
      Out[(size_t)(b * SEQ + row) * DIM + dcol] = acc[nt][r] * linv[r];
    }
  }
}

// ---------------------------------------------------------------------------
extern "C" void kernel_launch(void* const* d_in, const int* in_sizes, int n_in,
                              void* d_out, int out_size, void* d_ws,
                              size_t ws_size, hipStream_t stream) {
  const float* X    = (const float*)d_in[0];   // [4,4096,1024]
  const float* W    = (const float*)d_in[1];   // [3072,1024]
  const float* bias = (const float*)d_in[2];   // [3072]
  float* Out = (float*)d_out;                  // [4,4096,1024]

  // Workspace layout (f16): Qh, Kh [ROWS][DIM]; Vt [B][DIM][SEQ];
  // optional Xh [ROWS][DIM] and Wh [NE][DIM] for the all-f16 GEMM path.
  _Float16* Qh = (_Float16*)d_ws;
  _Float16* Kh = Qh + (size_t)ROWS * DIM;
  _Float16* Vt = Kh + (size_t)ROWS * DIM;
  _Float16* Xh = Vt + (size_t)ROWS * DIM;
  _Float16* Wh = Xh + (size_t)ROWS * DIM;

  const size_t need_ext =
      ((size_t)3 * ROWS * DIM + (size_t)ROWS * DIM + (size_t)NE * DIM) *
      sizeof(_Float16);                        // ~140.5 MB

  if (ws_size >= need_ext) {
    // One-time f32->f16 conversion of X and W (memory-bound, ~5 us).
    const int nx = ROWS * DIM;                 // 16,777,216
    const int nw = NE * DIM;                   //  3,145,728
    cvt_f32_f16_kernel<<<nx / (256 * 8), 256, 0, stream>>>(X, Xh, nx);
    cvt_f32_f16_kernel<<<nw / (256 * 8), 256, 0, stream>>>(W, Wh, nw);
    dim3 g1(NE / 64, ROWS / 32);               // 48 x 512 one-wave tiles
    qkv_f16_kernel<<<g1, 32, 0, stream>>>(Xh, Wh, bias, Qh, Kh, Vt);
  } else {
    dim3 g1(NE / 64, ROWS / 16);               // 48 x 1024 one-wave tiles
    qkv_f32_kernel<<<g1, 32, 0, stream>>>(X, W, bias, Qh, Kh, Vt);
  }

  attn_kernel<<<ROWS / 16, 128, 0, stream>>>(Qh, Kh, Vt, Out);
}
